// VLSTMModel_3882650435795
// MI455X (gfx1250) — matrix operand order repaired and checked by reading, not compiled, not run on hardware
//
#include <hip/hip_runtime.h>
#include <hip/hip_bf16.h>
#include <stdint.h>

#define S_   64
#define N_   65536
#define IN_  2
#define EMB_ 64
#define R_   128
#define OUT_ 5

// ---- padded LDS row strides (bank-conflict-free tile loads) ----
#define WIH_STRIDE 144   // 64 bf16 cols + 16B pad  (9 x 16B)
#define ROW_STRIDE 272   // 128 bf16 cols + 16B pad (17 x 16B)

// ---- LDS image layout (weights prepacked in d_ws with identical layout) ----
// Wih/Whh rows are stored j-major with gates interleaved:
//   image_row = j*64 + g*16 + r   (j: hidden col tile, g: gate, r: row in tile)
// so each j-iteration addresses a contiguous <18KB span -> DS immediate offsets.
#define OFF_WHH   73728            // 512*144
#define OFF_WOUT  212992           // +512*272
#define OFF_WEND  217344           // +16*272
#define OFF_BIAS  217344           // 512 f32
#define OFF_WEMB  219392           // 64x2 f32
#define OFF_BEMB  219904           // 64 f32
#define OFF_BOUT  220160           // 16 f32
#define OFF_H     220224           // 8 waves * 16 * 272
#define OFF_EHN   255040           // 8 waves * 16 * 272  (e and h_new, time-disjoint)
#define SMEM_TOTAL 289856

typedef __bf16   v16bf  __attribute__((ext_vector_type(16)));
typedef float    v8f    __attribute__((ext_vector_type(8)));
typedef uint32_t uint4v __attribute__((ext_vector_type(4)));
typedef int      v4i_   __attribute__((vector_size(4 * sizeof(int))));

union TileU { struct { uint4v lo, hi; } u; v16bf v; };

#if defined(__HIP_DEVICE_COMPILE__) && __has_builtin(__builtin_amdgcn_global_load_async_to_lds_b128)
#define USE_ASYNC_LDS 1
#else
#define USE_ASYNC_LDS 0
#endif

// A-operand (16x32 bf16): lanes 0-15: VGPR0-3=K0..7, VGPR4-7=K16..23;
// lanes 16-31: K8..15 / K24..31  -> two b128 at +0/+32 with 16B half-wave bias.
__device__ __forceinline__ v16bf load_opnd_a(const void* base, int rowStrideB,
                                             int lrow, int kc, int hsel) {
  const char* p = (const char*)base + lrow * rowStrideB + kc * 64 + hsel * 16;
  TileU t;
  t.u.lo = *(const uint4v*)(p);
  t.u.hi = *(const uint4v*)(p + 32);
  return t.v;
}

// B-operand (32x16 bf16): lanes 0-15: K0..7 / K16..23; lanes 16-31: K8..15 / K24..31
// -> two b128 at +0/+32 with 16B half-wave bias (matches A).
__device__ __forceinline__ v16bf load_opnd_b(const void* base, int rowStrideB,
                                             int lcol, int kc, int hsel) {
  const char* p = (const char*)base + lcol * rowStrideB + kc * 64 + hsel * 16;
  TileU t;
  t.u.lo = *(const uint4v*)(p);
  t.u.hi = *(const uint4v*)(p + 32);
  return t.v;
}

__device__ __forceinline__ v8f wmma_bf16(v16bf a, v16bf b, v8f c) {
  return __builtin_amdgcn_wmma_f32_16x16x32_bf16(
      false, a, false, b, (short)0, c, false, false);
}

// ---- activations: use CDNA5 hardware V_TANH_F32 when the builtin exists ----
__device__ __forceinline__ float tanh_ref(float x) {
  float t = __expf(-2.0f * fabsf(x));
  float r = (1.0f - t) / (1.0f + t);
  return x < 0.0f ? -r : r;
}
#if defined(__HIP_DEVICE_COMPILE__) && __has_builtin(__builtin_amdgcn_tanhf)
#define HW_TANH(x) __builtin_amdgcn_tanhf(x)
#elif defined(__HIP_DEVICE_COMPILE__) && __has_builtin(__builtin_amdgcn_tanh_f32)
#define HW_TANH(x) __builtin_amdgcn_tanh_f32(x)
#else
#define HW_TANH(x) tanh_ref(x)
#endif
// sigmoid(x) = 0.5*tanh(x/2) + 0.5  -> mul + tanh + fma
__device__ __forceinline__ float sigm(float x) {
  return fmaf(0.5f, HW_TANH(0.5f * x), 0.5f);
}

// ---- pre-pass: emit bf16 weight image in padded, j-major/gate-interleaved
// LDS layout ----
__global__ void convert_weights(const float* __restrict__ Wih,
                                const float* __restrict__ Whh,
                                const float* __restrict__ Wout,
                                char* __restrict__ img) {
  const int WIH_H  = 512 * 72;   // halves incl. pad
  const int WHH_H  = 512 * 136;
  const int WOUT_H = 16 * 136;
  int i = blockIdx.x * blockDim.x + threadIdx.x;
  if (i >= WIH_H + WHH_H + WOUT_H) return;
  __bf16 v; int off;
  if (i < WIH_H) {
    int rn = i / 72, c = i % 72;
    int jj = rn >> 6, g = (rn >> 4) & 3, r = rn & 15;
    int cc = c < EMB_ ? c : EMB_ - 1;
    float f = Wih[(g * 128 + jj * 16 + r) * EMB_ + cc];
    v = (c < EMB_) ? (__bf16)f : (__bf16)0.0f;
    off = i;
  } else if (i < WIH_H + WHH_H) {
    int k = i - WIH_H, rn = k / 136, c = k % 136;
    int jj = rn >> 6, g = (rn >> 4) & 3, r = rn & 15;
    int cc = c < R_ ? c : R_ - 1;
    float f = Whh[(g * 128 + jj * 16 + r) * R_ + cc];
    v = (c < R_) ? (__bf16)f : (__bf16)0.0f;
    off = WIH_H + k;
  } else {
    int k = i - WIH_H - WHH_H, r = k / 136, c = k % 136;
    int cc = c < R_ ? c : R_ - 1;
    int rr = r < OUT_ ? r : 0;
    float f = Wout[rr * R_ + cc];
    v = (r < OUT_ && c < R_) ? (__bf16)f : (__bf16)0.0f;
    off = WIH_H + WHH_H + k;
  }
  ((__bf16*)img)[off] = v;
}

// ---- main kernel: 256 threads = 8 waves, each wave owns 16 batch rows for
// the whole sequence; all weights resident in LDS; c in regs, h in LDS bf16.
__global__ __launch_bounds__(256) void vlstm_kernel(
    const float* __restrict__ input,   // S,N,2
    const float* __restrict__ h0,      // N,128
    const float* __restrict__ c0,      // N,128
    const int*   __restrict__ mask,    // S,N
    const float* __restrict__ Wemb,    // 64x2
    const float* __restrict__ bemb,    // 64
    const float* __restrict__ bih,     // 512
    const float* __restrict__ bhh,     // 512
    const float* __restrict__ bout,    // 5
    const char*  __restrict__ wimg,    // packed bf16 weight image (217344 B)
    float* __restrict__ out,           // S,N,5
    float* __restrict__ hf,            // N,128
    float* __restrict__ cf) {          // N,128
  extern __shared__ char smem[];
  char*  Wih_s  = smem;               // stride 144, j-major
  char*  Whh_s  = smem + OFF_WHH;     // stride 272, j-major
  char*  Wout_s = smem + OFF_WOUT;    // stride 272
  float* bias_s = (float*)(smem + OFF_BIAS);
  float* wemb_s = (float*)(smem + OFF_WEMB);
  float* bemb_s = (float*)(smem + OFF_BEMB);
  float* bout_s = (float*)(smem + OFF_BOUT);

  const int tid  = threadIdx.x;
  const int wave = tid >> 5;
  const int lane = tid & 31;
  const int lrow = lane & 15;
  const int hsel = (lane >> 4) & 1;
  const int rb   = blockIdx.x * 128 + wave * 16;

  char* h_w   = smem + OFF_H   + wave * (16 * ROW_STRIDE);
  char* ehn_w = smem + OFF_EHN + wave * (16 * ROW_STRIDE);

  // ---- bulk-copy the weight image into LDS (async-to-LDS when available) ----
  for (int i = tid; i < OFF_WEND / 16; i += 256) {
    const char* src = wimg + i * 16;
    char*       dst = smem + i * 16;
#if USE_ASYNC_LDS
    __builtin_amdgcn_global_load_async_to_lds_b128(
        (__attribute__((address_space(1))) v4i_*)(src),
        (__attribute__((address_space(3))) v4i_*)(dst), 0, 0);
#else
    *(uint4v*)dst = *(const uint4v*)src;
#endif
  }
  // small constants
  for (int i = tid; i < 4 * R_; i += 256) bias_s[i] = bih[i] + bhh[i];
  for (int i = tid; i < EMB_ * 2; i += 256) wemb_s[i] = Wemb[i];
  for (int i = tid; i < EMB_; i += 256) bemb_s[i] = bemb[i];
  if (tid < 16) bout_s[tid] = (tid < OUT_) ? bout[tid] : 0.0f;

  // ---- initial states (per-wave LDS region; c in registers, D-layout) ----
  float c_reg[8][8];
#pragma unroll
  for (int j = 0; j < 8; ++j) {
    int col = j * 16 + lrow;
#pragma unroll
    for (int vi = 0; vi < 8; ++vi)
      c_reg[j][vi] = c0[(size_t)(rb + vi + hsel * 8) * R_ + col];
  }
  {
    __bf16* hrow = (__bf16*)(h_w + lrow * ROW_STRIDE);
    for (int cc = 0; cc < 64; ++cc) {
      int col = hsel * 64 + cc;
      hrow[col] = (__bf16)h0[(size_t)(rb + lrow) * R_ + col];
    }
  }
#if USE_ASYNC_LDS
#if __has_builtin(__builtin_amdgcn_s_wait_asynccnt)
  __builtin_amdgcn_s_wait_asynccnt(0);
#else
  asm volatile("s_wait_asynccnt 0" ::: "memory");
#endif
#endif
  __syncthreads();

  // time-invariant per-lane constants (static indices after unroll)
  float bias_reg[8][4];
#pragma unroll
  for (int j = 0; j < 8; ++j) {
    int col = j * 16 + lrow;
#pragma unroll
    for (int g = 0; g < 4; ++g) bias_reg[j][g] = bias_s[g * R_ + col];
  }
  const float bq = bout_s[lrow];

#pragma unroll 1
  for (int t = 0; t < S_; ++t) {
    // ---- embedding: e = relu(x @ Wemb^T + bemb), K=2 -> VALU ----
    const size_t xb = ((size_t)t * N_ + rb + lrow) * IN_;
    float x0 = input[xb + 0], x1 = input[xb + 1];
    if (t + 1 < S_) __builtin_prefetch(&input[xb + (size_t)N_ * IN_], 0, 0);
    {
      __bf16* erow = (__bf16*)(ehn_w + lrow * ROW_STRIDE);
      int j0 = hsel * 32;
#pragma unroll
      for (int j = 0; j < 32; ++j) {
        int jj = j0 + j;
        float v = fmaf(x1, wemb_s[2 * jj + 1], fmaf(x0, wemb_s[2 * jj], bemb_s[jj]));
        erow[jj] = (__bf16)(v > 0.0f ? v : 0.0f);
      }
    }

    // ---- hoist A operands (e: K=64 in 2 chunks; h: K=128 in 4 chunks) ----
    v16bf ea[2], ha[4];
#pragma unroll
    for (int kc = 0; kc < 2; ++kc)
      ea[kc] = load_opnd_a(ehn_w, ROW_STRIDE, lrow, kc, hsel);
#pragma unroll
    for (int kc = 0; kc < 4; ++kc)
      ha[kc] = load_opnd_a(h_w, ROW_STRIDE, lrow, kc, hsel);

    const int mbase = t * N_ + rb + hsel * 8;
    int m8[8];
#pragma unroll
    for (int vi = 0; vi < 8; ++vi) m8[vi] = mask[mbase + vi];

    // ---- gates per hidden-column tile: 4 gates x (2+4) WMMAs, B from LDS.
    // Fully unrolled so c_reg/bias_reg stay register-resident.
#pragma unroll
    for (int j = 0; j < 8; ++j) {
      const char* wi_j = Wih_s + (size_t)j * (64 * WIH_STRIDE);
      const char* wh_j = Whh_s + (size_t)j * (64 * ROW_STRIDE);
      v8f acc[4];
#pragma unroll
      for (int g = 0; g < 4; ++g) {
        acc[g] = (v8f){0.f, 0.f, 0.f, 0.f, 0.f, 0.f, 0.f, 0.f};
        const char* wi = wi_j + g * (16 * WIH_STRIDE);
#pragma unroll
        for (int kc = 0; kc < 2; ++kc)
          acc[g] = wmma_bf16(ea[kc], load_opnd_b(wi, WIH_STRIDE, lrow, kc, hsel), acc[g]);
        const char* wh = wh_j + g * (16 * ROW_STRIDE);
#pragma unroll
        for (int kc = 0; kc < 4; ++kc)
          acc[g] = wmma_bf16(ha[kc], load_opnd_b(wh, ROW_STRIDE, lrow, kc, hsel), acc[g]);
      }
      // ---- LSTM cell elementwise (+bias) + masked state update ----
      int col = j * 16 + lrow;
#pragma unroll
      for (int vi = 0; vi < 8; ++vi) {
        float iv = sigm(acc[0][vi] + bias_reg[j][0]);
        float fv = sigm(acc[1][vi] + bias_reg[j][1]);
        float gv = HW_TANH(acc[2][vi] + bias_reg[j][2]);
        float ov = sigm(acc[3][vi] + bias_reg[j][3]);
        float cn = fv * c_reg[j][vi] + iv * gv;
        float hn = ov * HW_TANH(cn);
        int row = vi + hsel * 8;
        *(__bf16*)(ehn_w + row * ROW_STRIDE + col * 2) = (__bf16)hn;  // h_new staging
        if (m8[vi]) {
          c_reg[j][vi] = cn;
          *(__bf16*)(h_w + row * ROW_STRIDE + col * 2) = (__bf16)hn;  // recurrent h
        }
      }
    }

    // ---- output projection: 16x16 (K=128), W_out zero-padded to 16 rows ----
    v8f oacc = (v8f){0.f, 0.f, 0.f, 0.f, 0.f, 0.f, 0.f, 0.f};
#pragma unroll
    for (int kc = 0; kc < 4; ++kc)
      oacc = wmma_bf16(load_opnd_a(ehn_w, ROW_STRIDE, lrow, kc, hsel),
                       load_opnd_b(Wout_s, ROW_STRIDE, lrow, kc, hsel), oacc);
    if (lrow < OUT_) {
#pragma unroll
      for (int vi = 0; vi < 8; ++vi) {
        int row = vi + hsel * 8;
        out[((size_t)t * N_ + rb + row) * OUT_ + lrow] = m8[vi] ? oacc[vi] + bq : 0.0f;
      }
    }
  }

  // ---- final states ----
  {
    const __bf16* hrow = (const __bf16*)(h_w + lrow * ROW_STRIDE);
    for (int cc = 0; cc < 64; ++cc) {
      int col = hsel * 64 + cc;
      hf[(size_t)(rb + lrow) * R_ + col] = (float)hrow[col];
    }
  }
#pragma unroll
  for (int j = 0; j < 8; ++j) {
    int col = j * 16 + lrow;
#pragma unroll
    for (int vi = 0; vi < 8; ++vi)
      cf[(size_t)(rb + vi + hsel * 8) * R_ + col] = c_reg[j][vi];
  }
}

extern "C" void kernel_launch(void* const* d_in, const int* in_sizes, int n_in,
                              void* d_out, int out_size, void* d_ws, size_t ws_size,
                              hipStream_t stream) {
  const float* input = (const float*)d_in[0];
  const float* h0    = (const float*)d_in[1];
  const float* c0    = (const float*)d_in[2];
  const int*   maskp = (const int*)d_in[3];
  const float* Wemb  = (const float*)d_in[4];
  const float* bemb  = (const float*)d_in[5];
  const float* Wih   = (const float*)d_in[6];
  const float* Whh   = (const float*)d_in[7];
  const float* bih   = (const float*)d_in[8];
  const float* bhh   = (const float*)d_in[9];
  const float* Wout  = (const float*)d_in[10];
  const float* bout  = (const float*)d_in[11];

  char* wimg = (char*)d_ws;  // 217,344 B bf16 weight image (padded LDS layout)

  float* outp = (float*)d_out;
  float* hf   = outp + (size_t)S_ * N_ * OUT_;
  float* cf   = hf + (size_t)N_ * R_;

  const int halves = 512 * 72 + 512 * 136 + 16 * 136;  // 108,672
  hipLaunchKernelGGL(convert_weights, dim3((halves + 255) / 256), dim3(256), 0,
                     stream, Wih, Whh, Wout, wimg);
  hipLaunchKernelGGL(vlstm_kernel, dim3(N_ / 128), dim3(256), SMEM_TOTAL, stream,
                     input, h0, c0, maskp, Wemb, bemb, bih, bhh, bout,
                     wimg, outp, hf, cf);
}
